// RingDilatedAttentionHybridFixed_4784593568286
// MI455X (gfx1250) — matrix operand order repaired and checked by reading, not compiled, not run on hardware
//
#include <hip/hip_runtime.h>
#include <math.h>

typedef __attribute__((ext_vector_type(16))) _Float16 v16h;
typedef __attribute__((ext_vector_type(8)))  float    v8f;

namespace {
constexpr int SEQ   = 4096;
constexpr int HEADS = 16;
constexpr int DDIM  = 64;
constexpr int TOK   = HEADS * DDIM;   // 1024 floats per token row
constexpr int SEG   = 2048;           // only the last (seg,dil) group survives
constexpr int DIL   = 4;
constexpr int PSTR  = 36;             // P-staging row stride (f16 units)
constexpr int KSTR  = 66;             // K f16 row stride (64 + pad)
constexpr int VSTR  = 34;             // V f16 transposed row stride (32 + pad)
}

#if __has_builtin(__builtin_amdgcn_tensor_load_to_lds)
#define HAVE_TDM 1
#else
#define HAVE_TDM 0
#endif

#if __has_builtin(__builtin_amdgcn_global_load_async_to_lds_b128)
#define HAVE_ASYNC 1
#else
#define HAVE_ASYNC 0
#endif

__device__ __forceinline__ void cp16_to_lds(void* lds, const void* g) {
#if HAVE_ASYNC
  typedef int v4i_t __attribute__((ext_vector_type(4)));
  typedef __attribute__((address_space(1))) v4i_t* g_v4i_p;   // global src
  typedef __attribute__((address_space(3))) v4i_t* l_v4i_p;   // LDS dst
  __builtin_amdgcn_global_load_async_to_lds_b128((g_v4i_p)g, (l_v4i_p)lds, 0, 0);
#else
  *(float4*)lds = *(const float4*)g;
#endif
}

__device__ __forceinline__ void async_wait_all() {
#if HAVE_ASYNC
#if __has_builtin(__builtin_amdgcn_s_wait_asynccnt)
  __builtin_amdgcn_s_wait_asynccnt(0);
#else
  asm volatile("s_wait_asynccnt 0x0" ::: "memory");
#endif
#endif
}

#if HAVE_TDM
// 2D f32 tile: `rows` rows of `row_elems` elements, global row stride
// `row_stride_elems`, packed densely into LDS at byte offset lds_byte_off.
// D# layout per CDNA5 ISA ch.8 (group0: count/lds/global_addr/type=2,
// group1: data_size=4B, tensor_dim0/1, tile_dim0/1, dim0 stride).
__device__ __forceinline__ void tdm_load_tile_f32(const void* gsrc, unsigned lds_byte_off,
                                                  unsigned rows, unsigned row_elems,
                                                  unsigned row_stride_elems) {
  typedef unsigned __attribute__((ext_vector_type(4))) v4u;
  typedef int      __attribute__((ext_vector_type(4))) v4i;
  typedef int      __attribute__((ext_vector_type(8))) v8i;
  const unsigned long long ga = (unsigned long long)(size_t)gsrc;
  v4u g0;
  g0[0] = 1u;                                               // count=1 (user D#)
  g0[1] = lds_byte_off;                                     // lds_addr
  g0[2] = (unsigned)ga;                                     // global_addr[31:0]
  g0[3] = (unsigned)((ga >> 32) & 0x1FFFFFFull) | (2u << 30); // addr[56:32] | type=2
  v8i g1;
  g1[0] = 0x00020000;                                       // data_size=2 -> 4 bytes
  g1[1] = (int)((row_elems & 0xFFFFu) << 16);               // tensor_dim0[15:0]
  g1[2] = (int)((row_elems >> 16) | ((rows & 0xFFFFu) << 16)); // dim0 hi | dim1 lo
  g1[3] = (int)((rows >> 16) | ((row_elems & 0xFFFFu) << 16)); // dim1 hi | tile_dim0
  g1[4] = (int)(rows & 0xFFFFu);                            // tile_dim1 (tile_dim2=0)
  g1[5] = (int)row_stride_elems;                            // tensor_dim0_stride[31:0]
  g1[6] = 0;
  g1[7] = 0;
  v4i z4 = {0, 0, 0, 0};
#if defined(__clang_major__) && (__clang_major__ >= 23)
  v8i z8 = {0, 0, 0, 0, 0, 0, 0, 0};
  __builtin_amdgcn_tensor_load_to_lds(g0, g1, z4, z4, z8, 0);
#else
  __builtin_amdgcn_tensor_load_to_lds(g0, g1, z4, z4, 0);
#endif
}

__device__ __forceinline__ void tensor_wait_all() {
#if __has_builtin(__builtin_amdgcn_s_wait_tensorcnt)
  __builtin_amdgcn_s_wait_tensorcnt(0);
#else
  asm volatile("s_wait_tensorcnt 0x0" ::: "memory");
#endif
}
#endif  // HAVE_TDM

__global__ __launch_bounds__(128, 1)
void dilated_attn_wmma(const float* __restrict__ qg,
                       const float* __restrict__ kg,
                       const float* __restrict__ vg,
                       float* __restrict__ outg) {
  __shared__ __align__(16) float    Kf32[32 * DDIM];        // raw K tile (TDM dest)
  __shared__ __align__(16) float    Vf32[32 * DDIM];        // raw V tile (async dest)
  __shared__ __align__(16) _Float16 Kf16[32 * KSTR];        // K tile, f16 [key][d]
  __shared__ __align__(16) _Float16 Vf16[DDIM * VSTR];      // V tile, f16 [d][key]
  __shared__ __align__(16) _Float16 Pst[4][16 * PSTR];      // per-wave P transpose

  const int tid  = threadIdx.x;
  const int wave = tid >> 5;
  const int lane = tid & 31;
  const int half = lane >> 4;
  const int l16  = lane & 15;

  int bid = blockIdx.x;
  const int qblk = bid & 31;  bid >>= 5;
  const int h    = bid & 15;  bid >>= 4;
  const int nseg = bid;                       // segment 0 or 1

  const int segbase = nseg * SEG;
  const int qrow0   = segbase + qblk * 64 + wave * 16;
  const int vcount  = (SEQ - segbase) / DIL;  // 1024 / 512 valid keys
  const int ntiles  = vcount >> 5;

  // ---- Q tile (16x64) -> two f16 A operands, pre-scaled by 1/sqrt(64) ----
  v16h a_lo, a_hi;
  {
    const float* qp = qg + (size_t)(qrow0 + l16) * TOK + h * DDIM;
    #pragma unroll
    for (int e = 0; e < 8; ++e) {
      const int kk = (e < 4) ? (8 * half + 2 * e) : (16 + 8 * half + 2 * (e - 4));
      float2 f0 = *(const float2*)(qp + kk);
      float2 f1 = *(const float2*)(qp + 32 + kk);
      a_lo[2*e] = (_Float16)(f0.x * 0.125f); a_lo[2*e+1] = (_Float16)(f0.y * 0.125f);
      a_hi[2*e] = (_Float16)(f1.x * 0.125f); a_hi[2*e+1] = (_Float16)(f1.y * 0.125f);
    }
  }

  v8f o0 = {}, o1 = {}, o2 = {}, o3 = {};
  float m_run[8], l_run[8];
  #pragma unroll
  for (int e = 0; e < 8; ++e) { m_run[e] = -INFINITY; l_run[e] = 0.0f; }

  _Float16* pst = &Pst[wave][0];

  const int srow = tid >> 2;            // staging: 4 threads per key row
  const int soff = (tid & 3) * 16;      // 64B chunk within the 256B row

  for (int t = 0; t < ntiles; ++t) {
    const int keybase = t << 5;

    // ---- stage K tile: one TDM descriptor (32 rows x 256B, stride 16KB) ----
#if HAVE_TDM
    if (wave == 0) {
      tdm_load_tile_f32(kg + (size_t)(segbase + DIL * keybase) * TOK + h * DDIM,
                        (unsigned)(size_t)((__attribute__((address_space(3))) float*)&Kf32[0]),
                        32u, (unsigned)DDIM, (unsigned)(DIL * TOK));
    }
#else
    {
      const float* kp = kg + (size_t)(segbase + DIL * (keybase + srow)) * TOK + h * DDIM + soff;
      float* kd = &Kf32[srow * DDIM + soff];
      #pragma unroll
      for (int j = 0; j < 4; ++j) cp16_to_lds(kd + 4 * j, kp + 4 * j);
    }
#endif
    // ---- stage V tile: async global->LDS b128, all 128 threads ----
    {
      const float* vp = vg + (size_t)(segbase + DIL * (keybase + srow)) * TOK + h * DDIM + soff;
      float* vd = &Vf32[srow * DDIM + soff];
      #pragma unroll
      for (int j = 0; j < 4; ++j) cp16_to_lds(vd + 4 * j, vp + 4 * j);
    }
    async_wait_all();
#if HAVE_TDM
    if (wave == 0) tensor_wait_all();
#endif
    __syncthreads();

    // ---- convert once per block: K -> [key][d] f16, V -> [d][key] f16 ----
    {
      const int key = tid >> 2, d0 = (tid & 3) * 16;
      #pragma unroll
      for (int j = 0; j < 16; ++j)
        Kf16[key * KSTR + d0 + j] = (_Float16)Kf32[key * DDIM + d0 + j];
      const int dd = tid >> 1, kb = (tid & 1) * 16;
      #pragma unroll
      for (int j = 0; j < 16; ++j)
        Vf16[dd * VSTR + kb + j] = (_Float16)Vf32[(kb + j) * DDIM + dd];
    }
    __syncthreads();

    // ---- S = Q*K^T for 32 keys: operands as packed dword LDS reads ----
    v8f s0 = {}, s1 = {};
    #pragma unroll
    for (int j = 0; j < 2; ++j) {
      const int key = j * 16 + l16;
      union { v16h v; unsigned u[8]; } blo, bhi;
      #pragma unroll
      for (int e = 0; e < 8; ++e) {
        const int kk = (e < 4) ? (8 * half + 2 * e) : (16 + 8 * half + 2 * (e - 4));   // table K pattern
        blo.u[e] = *(const unsigned*)&Kf16[key * KSTR + kk];
        bhi.u[e] = *(const unsigned*)&Kf16[key * KSTR + 32 + kk];
      }
      v8f sj = {};
      sj = __builtin_amdgcn_wmma_f32_16x16x32_f16(false, a_lo, false, blo.v, (short)0, sj, false, false);
      sj = __builtin_amdgcn_wmma_f32_16x16x32_f16(false, a_hi, false, bhi.v, (short)0, sj, false, false);
      if (j == 0) s0 = sj; else s1 = sj;
    }

    // ---- online softmax ----
    float tmax[8];
    #pragma unroll
    for (int e = 0; e < 8; ++e) tmax[e] = fmaxf(s0[e], s1[e]);
    for (int off = 1; off < 16; off <<= 1) {
      #pragma unroll
      for (int e = 0; e < 8; ++e)
        tmax[e] = fmaxf(tmax[e], __shfl_xor(tmax[e], off, 32));
    }
    float fac[8];
    #pragma unroll
    for (int e = 0; e < 8; ++e) {
      const float mn = fmaxf(m_run[e], tmax[e]);
      fac[e] = __expf(m_run[e] - mn);
      m_run[e] = mn;
    }
    #pragma unroll
    for (int e = 0; e < 8; ++e) {
      const float p0 = __expf(s0[e] - m_run[e]);
      const float p1 = __expf(s1[e] - m_run[e]);
      l_run[e] = l_run[e] * fac[e] + p0 + p1;
      pst[(e + 8 * half) * PSTR + l16]      = (_Float16)p0;
      pst[(e + 8 * half) * PSTR + 16 + l16] = (_Float16)p1;
      o0[e] *= fac[e]; o1[e] *= fac[e]; o2[e] *= fac[e]; o3[e] *= fac[e];
    }
    __builtin_amdgcn_wave_barrier();   // DS ops are wave-ordered
    v16h pa;
    #pragma unroll
    for (int e = 0; e < 8; ++e) {
      const int kk = (e < 4) ? (8 * half + 2 * e) : (16 + 8 * half + 2 * (e - 4));
      pa[2*e]   = pst[l16 * PSTR + kk];
      pa[2*e+1] = pst[l16 * PSTR + kk + 1];
    }
    __builtin_amdgcn_wave_barrier();

    // ---- O += P * V : adjacent-key f16 pairs as packed dword LDS reads ----
    #pragma unroll
    for (int c = 0; c < 4; ++c) {
      union { v16h v; unsigned u[8]; } bv;
      #pragma unroll
      for (int e = 0; e < 8; ++e)
        bv.u[e] = *(const unsigned*)&Vf16[(c * 16 + l16) * VSTR + ((e < 4) ? (8 * half + 2 * e) : (16 + 8 * half + 2 * (e - 4)))];
      if      (c == 0) o0 = __builtin_amdgcn_wmma_f32_16x16x32_f16(false, pa, false, bv.v, (short)0, o0, false, false);
      else if (c == 1) o1 = __builtin_amdgcn_wmma_f32_16x16x32_f16(false, pa, false, bv.v, (short)0, o1, false, false);
      else if (c == 2) o2 = __builtin_amdgcn_wmma_f32_16x16x32_f16(false, pa, false, bv.v, (short)0, o2, false, false);
      else             o3 = __builtin_amdgcn_wmma_f32_16x16x32_f16(false, pa, false, bv.v, (short)0, o3, false, false);
    }
  }

  // ---- finalize: full row sums, normalize, store f32 ----
  for (int off = 1; off < 16; off <<= 1) {
    #pragma unroll
    for (int e = 0; e < 8; ++e)
      l_run[e] += __shfl_xor(l_run[e], off, 32);
  }
  #pragma unroll
  for (int e = 0; e < 8; ++e) {
    const float inv = 1.0f / l_run[e];
    const int row = qrow0 + e + 8 * half;
    float* op = outg + (size_t)row * TOK + h * DDIM + l16;
    op[0]  = o0[e] * inv;
    op[16] = o1[e] * inv;
    op[32] = o2[e] * inv;
    op[48] = o3[e] * inv;
  }
}

extern "C" void kernel_launch(void* const* d_in, const int* in_sizes, int n_in,
                              void* d_out, int out_size, void* d_ws, size_t ws_size,
                              hipStream_t stream) {
  const float* q = (const float*)d_in[0];
  const float* k = (const float*)d_in[1];
  const float* v = (const float*)d_in[2];
  float* out = (float*)d_out;
  (void)in_sizes; (void)n_in; (void)out_size; (void)d_ws; (void)ws_size;

  dim3 grid(2 * 16 * 32);   // nseg * heads * (2048/64) q-row blocks
  dim3 block(128);          // 4 wave32 waves, one 16-row q-tile each
  hipLaunchKernelGGL(dilated_attn_wmma, grid, block, 0, stream, q, k, v, out);
}
